// MultiHeadSelfAttention_50998441673162
// MI455X (gfx1250) — matrix operand order repaired and checked by reading, not compiled, not run on hardware
//
#include <hip/hip_runtime.h>
#include <hip/hip_bf16.h>

typedef __attribute__((ext_vector_type(16))) _Float16 v16h;
typedef __attribute__((ext_vector_type(8)))  _Float16 v8h;
typedef __attribute__((ext_vector_type(8)))  float    v8f;

#define DIMD   1024
#define HEADS  16
#define HD     64
#define LSEQ   2048
#define NTOK   4096          // B * L
#define QSCALE 0.125f        // HD^-0.5
#define LSCALE 0.5f          // alpha / rank

// ---------------------------------------------------------------- cvt f32->f16
__global__ void cvt_f32_f16(const float* __restrict__ in, _Float16* __restrict__ out, int n) {
    int i = blockIdx.x * blockDim.x + threadIdx.x;
    if (i < n) out[i] = (_Float16)in[i];
}

// -------------------------------------------------- fold LoRA into weight, f16
// out[j*1024+i] = W[j][i] + 0.5 * sum_r A[i][r] * B[r][j]   (layout [out][in])
__global__ void fold_w(const float* __restrict__ W, const float* __restrict__ A,
                       const float* __restrict__ Bm, _Float16* __restrict__ out) {
    int idx = blockIdx.x * 256 + threadIdx.x;      // 1M threads
    int j = idx >> 10, i = idx & 1023;
    float acc = 0.f;
#pragma unroll
    for (int r = 0; r < 16; ++r) acc += A[i * 16 + r] * Bm[r * DIMD + j];
    out[idx] = (_Float16)(W[idx] + LSCALE * acc);
}

// ---------------------------------------------------------------- GEMM (WMMA)
// X [4096 x 1024] f16 row-major; Wt [1024(out) x 1024(in)] f16 (B^T layout).
// mode 0: f16 -> [B,H,L,hd] (value scaled by scl)
// mode 1: f16 -> [B,H,hd,L] (transposed, for V)
// mode 2: f32 -> token-major [4096 x 1024] (final output)
__global__ __launch_bounds__(256)
void lora_gemm(const _Float16* __restrict__ X, const _Float16* __restrict__ Wt,
               const float* __restrict__ bias, _Float16* __restrict__ outH,
               float* __restrict__ outF, int mode, float scl) {
    const int lane = threadIdx.x & 31;
    const int wave = threadIdx.x >> 5;
    const int wy = wave >> 2, wx = wave & 3;
    const int m0 = blockIdx.x * 128 + wy * 64;
    const int n0 = blockIdx.y * 128 + wx * 32;
    const int r16 = lane & 15, g = lane >> 4;

    v8f c[4][2] = {};
    for (int k0 = 0; k0 < DIMD; k0 += 32) {
        v16h a[4], b[2];
#pragma unroll
        for (int mt = 0; mt < 4; ++mt) {           // A frag: lane=m, two 8-K runs
            const _Float16* p = X + (long)(m0 + mt * 16 + r16) * DIMD + k0 + g * 8;
            v8h lo = *(const v8h*)p;
            v8h hi = *(const v8h*)(p + 16);
            a[mt] = __builtin_shufflevector(lo, hi, 0,1,2,3,4,5,6,7,8,9,10,11,12,13,14,15);
        }
#pragma unroll
        for (int nt = 0; nt < 2; ++nt) {           // B frag: lane=n, two 8-K runs
            const _Float16* p = Wt + (long)(n0 + nt * 16 + r16) * DIMD + k0 + g * 8;
            v8h lo = *(const v8h*)p;
            v8h hi = *(const v8h*)(p + 16);
            b[nt] = __builtin_shufflevector(lo, hi, 0,1,2,3,4,5,6,7,8,9,10,11,12,13,14,15);
        }
#pragma unroll
        for (int mt = 0; mt < 4; ++mt)
#pragma unroll
            for (int nt = 0; nt < 2; ++nt)
                c[mt][nt] = __builtin_amdgcn_wmma_f32_16x16x32_f16(
                    false, a[mt], false, b[nt], (short)0, c[mt][nt], false, false);
    }

#pragma unroll
    for (int mt = 0; mt < 4; ++mt)
#pragma unroll
        for (int nt = 0; nt < 2; ++nt) {
            int jn = n0 + nt * 16 + r16;
            float bj = bias[jn];
#pragma unroll
            for (int v = 0; v < 8; ++v) {
                int t = m0 + mt * 16 + v + 8 * g;          // token row
                float val = c[mt][nt][v] + bj;
                if (mode == 2) {
                    outF[(long)t * DIMD + jn] = val;
                } else {
                    int bb = t >> 11, l = t & 2047;
                    int h = jn >> 6,  d = jn & 63;
                    _Float16 hv = (_Float16)(val * scl);
                    if (mode == 0)
                        outH[(((long)(bb * HEADS + h) * LSEQ) + l) * HD + d] = hv;
                    else
                        outH[(((long)(bb * HEADS + h) * HD) + d) * LSEQ + l] = hv;
                }
            }
        }
}

// ------------------------------------------------------------- flash attention
__device__ __forceinline__ float redmax16(float x) {
#pragma unroll
    for (int m = 1; m < 16; m <<= 1) x = fmaxf(x, __shfl_xor(x, m, 16));
    return x;
}
__device__ __forceinline__ float redsum16(float x) {
#pragma unroll
    for (int m = 1; m < 16; m <<= 1) x += __shfl_xor(x, m, 16);
    return x;
}

__global__ __launch_bounds__(256)
void attn_kernel(const _Float16* __restrict__ Qh, const _Float16* __restrict__ Kh,
                 const _Float16* __restrict__ Vth, const float* __restrict__ bias,
                 _Float16* __restrict__ AOh) {
    __shared__ _Float16 lds[8 * 16 * 32];          // 1 KB per wave
    const int lane = threadIdx.x & 31;
    const int wave = threadIdx.x >> 5;
    const int r16 = lane & 15, g = lane >> 4;
    const int bh = blockIdx.x;                     // b*16 + h
    const int q0 = blockIdx.y * 128 + wave * 16;   // 16 q-rows per wave

    // Q A-fragments for K(dim)=0..31 and 32..63
    v16h qa[2];
#pragma unroll
    for (int kk = 0; kk < 2; ++kk) {
        const _Float16* p = Qh + ((long)bh * LSEQ + q0 + r16) * HD + kk * 32 + g * 8;
        v8h lo = *(const v8h*)p;
        v8h hi = *(const v8h*)(p + 16);
        qa[kk] = __builtin_shufflevector(lo, hi, 0,1,2,3,4,5,6,7,8,9,10,11,12,13,14,15);
    }

    float rm[8], rl[8];
#pragma unroll
    for (int v = 0; v < 8; ++v) { rm[v] = -1e30f; rl[v] = 0.f; }
    v8f o[4] = {};

    const float* bp = bias + ((long)(bh & 15) * LSEQ + q0) * LSEQ;
    _Float16* lp = lds + wave * 512;

    for (int key0 = 0; key0 < LSEQ; key0 += 32) {
        // scores S = Q K^T (+bias); two 16-key C tiles
        v8f s[2];
#pragma unroll
        for (int j = 0; j < 2; ++j) {
            const _Float16* kp = Kh + ((long)bh * LSEQ + key0 + j * 16 + r16) * HD + g * 8;
            v16h bk0 = __builtin_shufflevector(*(const v8h*)kp, *(const v8h*)(kp + 16), 0,1,2,3,4,5,6,7,8,9,10,11,12,13,14,15);           // dims 0..31
            v16h bk1 = __builtin_shufflevector(*(const v8h*)(kp + 32), *(const v8h*)(kp + 48), 0,1,2,3,4,5,6,7,8,9,10,11,12,13,14,15);    // dims 32..63
            v8f z = {};
            s[j] = __builtin_amdgcn_wmma_f32_16x16x32_f16(false, qa[0], false, bk0, (short)0, z,    false, false);
            s[j] = __builtin_amdgcn_wmma_f32_16x16x32_f16(false, qa[1], false, bk1, (short)0, s[j], false, false);
#pragma unroll
            for (int v = 0; v < 8; ++v)            // streamed bias: non-temporal
                s[j][v] += __builtin_nontemporal_load(
                    bp + (long)(v + 8 * g) * LSEQ + key0 + j * 16 + r16);
        }
        // online softmax over 32 keys
        v8f tmax;
#pragma unroll
        for (int v = 0; v < 8; ++v) tmax[v] = redmax16(fmaxf(s[0][v], s[1][v]));
        float fs[8];
#pragma unroll
        for (int v = 0; v < 8; ++v) {
            float nm = fmaxf(rm[v], tmax[v]);
            fs[v] = __expf(rm[v] - nm);
            rm[v] = nm;
        }
        v8f p0, p1;
#pragma unroll
        for (int v = 0; v < 8; ++v) {
            p0[v] = __expf(s[0][v] - rm[v]);
            p1[v] = __expf(s[1][v] - rm[v]);
            rl[v] = rl[v] * fs[v] + redsum16(p0[v] + p1[v]);
        }
#pragma unroll
        for (int nt = 0; nt < 4; ++nt)
#pragma unroll
            for (int v = 0; v < 8; ++v) o[nt][v] *= fs[v];

        // C-layout P -> LDS -> A-layout fragment
#pragma unroll
        for (int v = 0; v < 8; ++v) {
            lp[(v + 8 * g) * 32 + r16]      = (_Float16)p0[v];
            lp[(v + 8 * g) * 32 + 16 + r16] = (_Float16)p1[v];
        }
        asm volatile("s_wait_dscnt 0" ::: "memory");
        v16h pa;
        {
            const _Float16* p = lp + r16 * 32 + g * 8;
            v8h lo = *(const v8h*)p;
            v8h hi = *(const v8h*)(p + 16);
            pa = __builtin_shufflevector(lo, hi, 0,1,2,3,4,5,6,7,8,9,10,11,12,13,14,15);
        }
        asm volatile("" ::: "memory");
        // O += P @ V   (V transposed -> contiguous B^T rows)
#pragma unroll
        for (int nt = 0; nt < 4; ++nt) {
            const _Float16* vp = Vth + ((long)bh * HD + nt * 16 + r16) * LSEQ + key0 + g * 8;
            v16h vb = __builtin_shufflevector(*(const v8h*)vp, *(const v8h*)(vp + 16), 0,1,2,3,4,5,6,7,8,9,10,11,12,13,14,15);
            o[nt] = __builtin_amdgcn_wmma_f32_16x16x32_f16(false, pa, false, vb, (short)0, o[nt], false, false);
        }
    }

    // normalize and merge heads: token-major [4096 x 1024] f16
    const int b = bh >> 4, h = bh & 15;
#pragma unroll
    for (int nt = 0; nt < 4; ++nt)
#pragma unroll
        for (int v = 0; v < 8; ++v) {
            int row = q0 + v + 8 * g;
            int col = h * HD + nt * 16 + r16;
            AOh[((long)b * LSEQ + row) * DIMD + col] = (_Float16)(o[nt][v] / rl[v]);
        }
}

// ------------------------------------------------------------------- launcher
extern "C" void kernel_launch(void* const* d_in, const int* in_sizes, int n_in,
                              void* d_out, int out_size, void* d_ws, size_t ws_size,
                              hipStream_t stream) {
    const float* x     = (const float*)d_in[0];
    const float* abias = (const float*)d_in[1];
    const float* Wq = (const float*)d_in[2],  *bq = (const float*)d_in[3];
    const float* Aq = (const float*)d_in[4],  *Bq = (const float*)d_in[5];
    const float* Wk = (const float*)d_in[6],  *bk = (const float*)d_in[7];
    const float* Ak = (const float*)d_in[8],  *Bk = (const float*)d_in[9];
    const float* Wv = (const float*)d_in[10], *bv = (const float*)d_in[11];
    const float* Av = (const float*)d_in[12], *Bv = (const float*)d_in[13];
    const float* Wo = (const float*)d_in[14], *bo = (const float*)d_in[15];
    const float* Ao = (const float*)d_in[16], *Bo = (const float*)d_in[17];

    char* ws = (char*)d_ws;
    _Float16* xh  = (_Float16*)(ws);
    _Float16* wq  = (_Float16*)(ws + (8L  << 20));
    _Float16* wk  = (_Float16*)(ws + (10L << 20));
    _Float16* wv  = (_Float16*)(ws + (12L << 20));
    _Float16* wo  = (_Float16*)(ws + (14L << 20));
    _Float16* qh  = (_Float16*)(ws + (16L << 20));
    _Float16* kh  = (_Float16*)(ws + (24L << 20));
    _Float16* vth = (_Float16*)(ws + (32L << 20));
    _Float16* aoh = (_Float16*)(ws + (40L << 20));

    cvt_f32_f16<<<NTOK * DIMD / 256, 256, 0, stream>>>(x, xh, NTOK * DIMD);

    fold_w<<<DIMD * DIMD / 256, 256, 0, stream>>>(Wq, Aq, Bq, wq);
    fold_w<<<DIMD * DIMD / 256, 256, 0, stream>>>(Wk, Ak, Bk, wk);
    fold_w<<<DIMD * DIMD / 256, 256, 0, stream>>>(Wv, Av, Bv, wv);
    fold_w<<<DIMD * DIMD / 256, 256, 0, stream>>>(Wo, Ao, Bo, wo);

    dim3 ggrid(NTOK / 128, DIMD / 128);
    lora_gemm<<<ggrid, 256, 0, stream>>>(xh, wq, bq, qh,  nullptr, 0, QSCALE);
    lora_gemm<<<ggrid, 256, 0, stream>>>(xh, wk, bk, kh,  nullptr, 0, 1.0f);
    lora_gemm<<<ggrid, 256, 0, stream>>>(xh, wv, bv, vth, nullptr, 1, 1.0f);

    attn_kernel<<<dim3(2 * HEADS, LSEQ / 128), 256, 0, stream>>>(qh, kh, vth, abias, aoh);

    lora_gemm<<<ggrid, 256, 0, stream>>>(aoh, wo, bo, nullptr, (float*)d_out, 2, 1.0f);
}
